// Int8GPTJMLP_2662879724431
// MI455X (gfx1250) — 8-bit matrix operand order repaired and checked by reading, compiled, not run on hardware
//
#include <hip/hip_runtime.h>
#include <stdint.h>

typedef int v8i __attribute__((ext_vector_type(8)));

constexpr int N_TOK = 4096;
constexpr int E_DIM = 4096;
constexpr int I_DIM = 16384;

__device__ __forceinline__ v8i load_a_frag(const int8_t* __restrict__ p) {
    const int4 d0 = *(const int4*)(p +  0);
    const int4 d1 = *(const int4*)(p + 32);
    v8i a;
    a[0] = d0.x; a[1] = d0.y; a[2] = d0.z; a[3] = d0.w;
    a[4] = d1.x; a[5] = d1.y; a[6] = d1.z; a[7] = d1.w;
    return a;
}

__device__ __forceinline__ v8i load_b_frag(const int8_t* __restrict__ p) {
    const int4 lo = *(const int4*)(p +  0);
    const int4 hi = *(const int4*)(p + 32);
    v8i b;
    b[0] = lo.x; b[1] = lo.y; b[2] = lo.z; b[3] = lo.w;
    b[4] = hi.x; b[5] = hi.y; b[6] = hi.z; b[7] = hi.w;
    return b;
}

template <int KTOT>
__device__ __forceinline__ void gemm64x64(const int8_t* __restrict__ A, int lda,
                                          const int8_t* __restrict__ B, int ldb,
                                          int lane, v8i acc[4][4]) {
    const int lo16 = lane & 15;
    const int hiA  = (lane >> 4) << 4;
    const int hiB  = (lane >> 4) << 4;

    const int8_t* aP[4];
    const int8_t* bP[4];
#pragma unroll
    for (int i = 0; i < 4; ++i)
        aP[i] = A + (size_t)(i * 16 + lo16) * lda + hiA;
#pragma unroll
    for (int j = 0; j < 4; ++j)
        bP[j] = B + (size_t)(j * 16 + lo16) * ldb + hiB;

    for (int k0 = 0; k0 < KTOT; k0 += 64) {
        v8i af[4], bf[4];
#pragma unroll
        for (int i = 0; i < 4; ++i) {
            af[i] = load_a_frag(aP[i] + k0);
            __builtin_prefetch(aP[i] + k0 + 256, 0, 3);
        }
#pragma unroll
        for (int j = 0; j < 4; ++j) {
            bf[j] = load_b_frag(bP[j] + k0);
            __builtin_prefetch(bP[j] + k0 + 256, 0, 3);
        }
#pragma unroll
        for (int i = 0; i < 4; ++i)
#pragma unroll
            for (int j = 0; j < 4; ++j)
                acc[i][j] = __builtin_amdgcn_wmma_i32_16x16x64_iu8(
                     true, af[i],  true, bf[j],
                    acc[i][j],  false,  false);
    }
}

__device__ __forceinline__ void zero_acc(v8i acc[4][4]) {
#pragma unroll
    for (int i = 0; i < 4; ++i)
#pragma unroll
        for (int j = 0; j < 4; ++j)
#pragma unroll
            for (int r = 0; r < 8; ++r)
                acc[i][j][r] = 0;
}

__global__ __launch_bounds__(128)
void fc1_gelu_q8(const int8_t* __restrict__ x, const int8_t* __restrict__ W1,
                 const int8_t* __restrict__ b1,
                 const float* __restrict__ pa1, const float* __restrict__ pb1s,
                 int8_t* __restrict__ h) {
    const int lane = threadIdx.x & 31;
    const int wave = threadIdx.x >> 5;
    const int n0 = blockIdx.x * 256 + wave * 64;
    const int m0 = blockIdx.y * 64;
    const float a1  = *pa1;
    const float b1s = *pb1s;

    v8i acc[4][4];
    zero_acc(acc);
    gemm64x64<E_DIM>(x + (size_t)m0 * E_DIM, E_DIM,
                     W1 + (size_t)n0 * E_DIM, E_DIM, lane, acc);

    const int mrow = (lane >> 4) << 3;
    const int ncol = lane & 15;
#pragma unroll
    for (int j = 0; j < 4; ++j) {
        const int n = n0 + j * 16 + ncol;
        const float bias = (float)b1[n] * b1s;
#pragma unroll
        for (int i = 0; i < 4; ++i) {
#pragma unroll
            for (int r = 0; r < 8; ++r) {
                const int m = m0 + i * 16 + mrow + r;
                const float y = (float)acc[i][j][r] * a1 + bias;
                const float g = 0.5f * y * (1.0f + erff(y * 0.70710678118654752f));
                float q = rintf(g);
                q = fminf(fmaxf(q, -128.0f), 127.0f);
                h[(size_t)m * I_DIM + n] = (int8_t)(int)q;
            }
        }
    }
}

__global__ __launch_bounds__(128)
void fc2_f32(const int8_t* __restrict__ h, const int8_t* __restrict__ W2,
             const float* __restrict__ b2, const float* __restrict__ pa2,
             float* __restrict__ out) {
    const int lane = threadIdx.x & 31;
    const int wave = threadIdx.x >> 5;
    const int n0 = blockIdx.x * 256 + wave * 64;
    const int m0 = blockIdx.y * 64;
    const float a2 = *pa2;

    v8i acc[4][4];
    zero_acc(acc);
    gemm64x64<I_DIM>(h + (size_t)m0 * I_DIM, I_DIM,
                     W2 + (size_t)n0 * I_DIM, I_DIM, lane, acc);

    const int mrow = (lane >> 4) << 3;
    const int ncol = lane & 15;
#pragma unroll
    for (int j = 0; j < 4; ++j) {
        const int n = n0 + j * 16 + ncol;
        const float bias = b2[n];
#pragma unroll
        for (int i = 0; i < 4; ++i) {
#pragma unroll
            for (int r = 0; r < 8; ++r) {
                const int m = m0 + i * 16 + mrow + r;
                out[(size_t)m * E_DIM + n] = (float)acc[i][j][r] * a2 + bias;
            }
        }
    }
}

extern "C" void kernel_launch(void* const* d_in, const int* in_sizes, int n_in,
                              void* d_out, int out_size, void* d_ws, size_t ws_size,
                              hipStream_t stream) {
    const int8_t* x   = (const int8_t*)d_in[0];
    const int8_t* W1  = (const int8_t*)d_in[1];
    const int8_t* b1  = (const int8_t*)d_in[2];
    const float*  a1  = (const float*)d_in[3];
    const float*  b1s = (const float*)d_in[4];
    const int8_t* W2  = (const int8_t*)d_in[5];
    const float*  b2  = (const float*)d_in[6];
    const float*  a2  = (const float*)d_in[7];
    float* out = (float*)d_out;
    int8_t* h  = (int8_t*)d_ws;

    dim3 blk(128, 1, 1);

    dim3 g1(I_DIM / 256, N_TOK / 64, 1);
    fc1_gelu_q8<<<g1, blk, 0, stream>>>(x, W1, b1, a1, b1s, h);

    dim3 g2(E_DIM / 256, N_TOK / 64, 1);
    fc2_f32<<<g2, blk, 0, stream>>>(h, W2, b2, a2, out);
}
